// self_attention_12902081757879
// MI455X (gfx1250) — matrix operand order repaired and checked by reading, not compiled, not run on hardware
//
#include <hip/hip_runtime.h>

// Self-attention (SAGAN-style) for MI455X / gfx1250.
// Pipeline (all bf16 WMMA, f32 accum; attention matrix never materialized):
//   k0: convert weights f32 -> bf16
//   k1: projections  f=(N,64), gT=(N,64), hv=(C,N)  via v_wmma_f32_16x16x32_bf16
//   k2: softmax row stats m[i], 1/l[i]  (WMMA S tiles + shfl_xor reductions)
//   k3: out = gamma * (hv @ softmax(f@g)) + x, recomputing S tiles on the fly
// Workspace: ~20.8 MB (weights-bf16 640KB, f/gT 4MB, hv 16MB, stats 128KB).

#define BB   4
#define CC   512
#define C8V  64
#define NTOK 4096

typedef __attribute__((ext_vector_type(16))) __bf16 v16bf;
typedef __attribute__((ext_vector_type(8)))  float  v8f;

union FragU { v16bf v; uint4 q[2]; };
union Pack8 { uint4 q; unsigned short h[8]; };

__device__ __forceinline__ unsigned short f2bf_bits(float f) {
  union { float f; unsigned u; } v; v.f = f;
  unsigned r = (v.u + 0x7FFFu + ((v.u >> 16) & 1u)) >> 16;   // RNE
  return (unsigned short)r;
}

__device__ __forceinline__ v8f wmma_bf16(v16bf a, v16bf b, v8f c) {
  // D = A(16x32) * B(32x16) + C(16x16 f32)
  return __builtin_amdgcn_wmma_f32_16x16x32_bf16(false, a, false, b, (short)0, c, false, false);
}

// zero bf16 fragment
__device__ __forceinline__ v16bf z0_frag() {
  FragU f;
  f.q[0] = make_uint4(0, 0, 0, 0);
  f.q[1] = make_uint4(0, 0, 0, 0);
  return f.v;
}

// A fragment: rows row0..row0+15 (M), cols k0..k0+31 (K), row-major bf16 source.
// lanes 0-15: K = k0+0..7 (q0), k0+16..23 (q1); lanes 16-31: +8 / +24. (ISA 7.12.2)
__device__ __forceinline__ v16bf load_frag_a(const unsigned short* p, size_t rs,
                                             int row0, int k0, int lane) {
  int l16 = lane & 15, hi = lane >> 4;
  const unsigned short* rp = p + (size_t)(row0 + l16) * rs + k0 + hi * 8;
  FragU f;
  f.q[0] = *(const uint4*)(rp);
  f.q[1] = *(const uint4*)(rp + 16);
  return f.v;
}

// B fragment: 32x16 (KxN). Source stored as [n][k] row-major (K contiguous).
// lanes 0-15: K = k0+0..7 (q0), k0+16..23 (q1); lanes 16-31: +8 / +24.
__device__ __forceinline__ v16bf load_frag_b(const unsigned short* p, size_t rs,
                                             int row0, int k0, int lane) {
  int l16 = lane & 15, hi = lane >> 4;
  const unsigned short* rp = p + (size_t)(row0 + l16) * rs + k0 + hi * 8;
  FragU f;
  f.q[0] = *(const uint4*)(rp);
  f.q[1] = *(const uint4*)(rp + 16);
  return f.v;
}

// ---------------- k0: weight conversion ----------------
__global__ __launch_bounds__(256) void cvt_weights_kernel(
    const float* __restrict__ Wf, const float* __restrict__ Wg,
    const float* __restrict__ Wh,
    unsigned short* __restrict__ Wf_bf, unsigned short* __restrict__ Wg_bf,
    unsigned short* __restrict__ Wh_bf) {
  int i = blockIdx.x * 256 + threadIdx.x;
  if (i < C8V * CC) { Wf_bf[i] = f2bf_bits(Wf[i]); Wg_bf[i] = f2bf_bits(Wg[i]); }
  if (i < CC * CC)  { Wh_bf[i] = f2bf_bits(Wh[i]); }
}

// ---------------- k1: projections ----------------
// Grid (N/64, B), 512 threads (16 waves). Each WG owns 64 tokens.
// hv: all 16 waves (32 c-rows each). f: waves 0-3. gT: waves 4-7.
__global__ __launch_bounds__(512) void proj_kernel(
    const float* __restrict__ x,
    const float* __restrict__ bf, const float* __restrict__ bg,
    const float* __restrict__ bh,
    const unsigned short* __restrict__ Wf_bf,
    const unsigned short* __restrict__ Wg_bf,
    const unsigned short* __restrict__ Wh_bf,
    unsigned short* __restrict__ f_ws, unsigned short* __restrict__ gt_ws,
    unsigned short* __restrict__ hv_ws) {
  __shared__ __align__(16) unsigned short xT[64 * 40];  // [n][k], pad 40
  int tid = threadIdx.x, lane = tid & 31, w = tid >> 5;
  int l16 = lane & 15, hi = lane >> 4;
  int b = blockIdx.y, i0 = blockIdx.x * 64;

  v8f z = {0.f, 0.f, 0.f, 0.f, 0.f, 0.f, 0.f, 0.f};
  v8f acc[2][4], facc[4], gacc[4];
  for (int mt = 0; mt < 2; ++mt)
    for (int nt = 0; nt < 4; ++nt) acc[mt][nt] = z;
  for (int nt = 0; nt < 4; ++nt) { facc[nt] = z; gacc[nt] = z; }

  for (int k0 = 0; k0 < CC; k0 += 32) {
    // stage x[k0:k0+32, i0:i0+64] transposed into LDS as bf16 [n][k]
    for (int idx = tid; idx < 32 * 64; idx += 512) {
      int k = idx >> 6, n = idx & 63;
      float v = x[((size_t)(b * CC + k0 + k)) * NTOK + i0 + n];
      xT[n * 40 + k] = f2bf_bits(v);
    }
    __syncthreads();
    v16bf bx[4];
    for (int nt = 0; nt < 4; ++nt) bx[nt] = load_frag_b(xT, 40, 16 * nt, 0, lane);
    for (int mt = 0; mt < 2; ++mt) {
      v16bf a = load_frag_a(Wh_bf, CC, 32 * w + 16 * mt, k0, lane);
      for (int nt = 0; nt < 4; ++nt) acc[mt][nt] = wmma_bf16(a, bx[nt], acc[mt][nt]);
    }
    if (w < 4) {
      v16bf a = load_frag_a(Wf_bf, CC, 16 * w, k0, lane);
      for (int nt = 0; nt < 4; ++nt) facc[nt] = wmma_bf16(a, bx[nt], facc[nt]);
    } else if (w < 8) {
      v16bf a = load_frag_a(Wg_bf, CC, 16 * (w - 4), k0, lane);
      for (int nt = 0; nt < 4; ++nt) gacc[nt] = wmma_bf16(a, bx[nt], gacc[nt]);
    }
    __syncthreads();
  }
  // hv (C,N) bf16 store (+bias)
  for (int mt = 0; mt < 2; ++mt)
    for (int nt = 0; nt < 4; ++nt)
      for (int r = 0; r < 8; ++r) {
        int c = 32 * w + 16 * mt + r + 8 * hi;
        int n = i0 + 16 * nt + l16;
        hv_ws[((size_t)b * CC + c) * NTOK + n] = f2bf_bits(acc[mt][nt][r] + bh[c]);
      }
  if (w < 4) {  // f (N,64)
    for (int nt = 0; nt < 4; ++nt) {
      int n = i0 + 16 * nt + l16;
      Pack8 pk;
      for (int r = 0; r < 8; ++r) {
        int o = 16 * w + 8 * hi + r;
        pk.h[r] = f2bf_bits(facc[nt][r] + bf[o]);
      }
      *(uint4*)&f_ws[((size_t)b * NTOK + n) * 64 + 16 * w + 8 * hi] = pk.q;
    }
  } else if (w < 8) {  // gT (N,64): gt[n][o] = g[o][n]
    for (int nt = 0; nt < 4; ++nt) {
      int n = i0 + 16 * nt + l16;
      Pack8 pk;
      for (int r = 0; r < 8; ++r) {
        int o = 16 * (w - 4) + 8 * hi + r;
        pk.h[r] = f2bf_bits(gacc[nt][r] + bg[o]);
      }
      *(uint4*)&gt_ws[((size_t)b * NTOK + n) * 64 + 16 * (w - 4) + 8 * hi] = pk.q;
    }
  }
}

// ---------------- k2: softmax row stats ----------------
// Grid (N/256, B), 16 waves; wave owns 16 S-rows, sweeps all 4096 columns twice.
__global__ __launch_bounds__(512) void stats_kernel(
    const unsigned short* __restrict__ f_ws,
    const unsigned short* __restrict__ gt_ws,
    float* __restrict__ m_ws, float* __restrict__ rl_ws) {
  __shared__ __align__(16) unsigned short gs[16 * 64];  // one j-tile of gT
  int tid = threadIdx.x, lane = tid & 31, w = tid >> 5;
  int l16 = lane & 15, hi = lane >> 4;
  int b = blockIdx.y, irow = blockIdx.x * 256 + 16 * w;
  const unsigned short* fb = f_ws + (size_t)b * NTOK * 64;
  const unsigned short* gb = gt_ws + (size_t)b * NTOK * 64;
  v8f z = {0.f, 0.f, 0.f, 0.f, 0.f, 0.f, 0.f, 0.f};
  v16bf a0 = load_frag_a(fb, 64, irow, 0, lane);
  v16bf a1 = load_frag_a(fb, 64, irow, 32, lane);

  float vmax[8];
  for (int r = 0; r < 8; ++r) vmax[r] = -3.0e38f;
  for (int j0 = 0; j0 < NTOK; j0 += 16) {           // sweep 1: row max
    ((unsigned int*)gs)[tid] = ((const unsigned int*)(gb + (size_t)j0 * 64))[tid];
    __syncthreads();
    v16bf b0 = load_frag_b(gs, 64, 0, 0, lane);
    v16bf b1 = load_frag_b(gs, 64, 0, 32, lane);
    v8f s = wmma_bf16(a1, b1, wmma_bf16(a0, b0, z));
    for (int r = 0; r < 8; ++r) vmax[r] = fmaxf(vmax[r], s[r]);
    __syncthreads();
  }
  for (int off = 1; off < 16; off <<= 1)
    for (int r = 0; r < 8; ++r) vmax[r] = fmaxf(vmax[r], __shfl_xor(vmax[r], off, 16));

  float vsum[8];
  for (int r = 0; r < 8; ++r) vsum[r] = 0.f;
  for (int j0 = 0; j0 < NTOK; j0 += 16) {           // sweep 2: sum exp
    ((unsigned int*)gs)[tid] = ((const unsigned int*)(gb + (size_t)j0 * 64))[tid];
    __syncthreads();
    v16bf b0 = load_frag_b(gs, 64, 0, 0, lane);
    v16bf b1 = load_frag_b(gs, 64, 0, 32, lane);
    v8f s = wmma_bf16(a1, b1, wmma_bf16(a0, b0, z));
    for (int r = 0; r < 8; ++r) vsum[r] += __expf(s[r] - vmax[r]);
    __syncthreads();
  }
  for (int off = 1; off < 16; off <<= 1)
    for (int r = 0; r < 8; ++r) vsum[r] += __shfl_xor(vsum[r], off, 16);

  if (l16 == 0) {
    for (int r = 0; r < 8; ++r) {
      int row = irow + r + 8 * hi;
      m_ws[(size_t)b * NTOK + row]  = vmax[r];
      rl_ws[(size_t)b * NTOK + row] = 1.0f / vsum[r];
    }
  }
}

// ---------------- k3: output GEMM out = gamma * hv @ P + x ----------------
// Grid (N/64, B); WG owns out[:, j0:j0+64]. Per 32-row i-block: waves 0-7
// recompute P tile (32x64) -> LDS transposed [j][i] bf16; all 16 waves run
// the C=512 GEMM (wave w: c rows 32w..32w+31).
__global__ __launch_bounds__(512) void attn_out_kernel(
    const float* __restrict__ x, const float* __restrict__ gamma,
    const unsigned short* __restrict__ f_ws,
    const unsigned short* __restrict__ gt_ws,
    const unsigned short* __restrict__ hv_ws,
    const float* __restrict__ m_ws, const float* __restrict__ rl_ws,
    float* __restrict__ out) {
  __shared__ __align__(16) unsigned short pT[64 * 40];  // [j][i], pad 40
  int tid = threadIdx.x, lane = tid & 31, w = tid >> 5;
  int l16 = lane & 15, hi = lane >> 4;
  int b = blockIdx.y, j0 = blockIdx.x * 64;
  int ti = w >> 2, tj = w & 3;
  const unsigned short* fb = f_ws + (size_t)b * NTOK * 64;
  const unsigned short* gb = gt_ws + (size_t)b * NTOK * 64;
  const unsigned short* hb = hv_ws + (size_t)b * CC * NTOK;
  v8f z = {0.f, 0.f, 0.f, 0.f, 0.f, 0.f, 0.f, 0.f};
  v8f acc[2][4];
  for (int mt = 0; mt < 2; ++mt)
    for (int nt = 0; nt < 4; ++nt) acc[mt][nt] = z;

  v16bf gB0 = z0_frag(), gB1 = z0_frag();
  if (w < 8) {  // g fragments reused for every i-block
    gB0 = load_frag_b(gb, 64, j0 + 16 * tj, 0, lane);
    gB1 = load_frag_b(gb, 64, j0 + 16 * tj, 32, lane);
  }

  for (int i0 = 0; i0 < NTOK; i0 += 32) {
    if (w < 8) {  // phase A: P tile
      v16bf fa0 = load_frag_a(fb, 64, i0 + 16 * ti, 0, lane);
      v16bf fa1 = load_frag_a(fb, 64, i0 + 16 * ti, 32, lane);
      v8f s = wmma_bf16(fa1, gB1, wmma_bf16(fa0, gB0, z));
      for (int r = 0; r < 8; ++r) {
        int row = i0 + 16 * ti + r + 8 * hi;
        float p = __expf(s[r] - m_ws[(size_t)b * NTOK + row]) *
                  rl_ws[(size_t)b * NTOK + row];
        pT[(16 * tj + l16) * 40 + 16 * ti + r + 8 * hi] = f2bf_bits(p);
      }
    }
    __syncthreads();
    // phase B: out tiles; prefetch next hv i-block while we compute
    __builtin_prefetch(hb + (size_t)(32 * w) * NTOK + i0 + 32, 0, 1);
    v16bf ha0 = load_frag_a(hb, NTOK, 32 * w, i0, lane);
    v16bf ha1 = load_frag_a(hb, NTOK, 32 * w + 16, i0, lane);
    for (int nt = 0; nt < 4; ++nt) {
      v16bf pB = load_frag_b(pT, 40, 16 * nt, 0, lane);
      acc[0][nt] = wmma_bf16(ha0, pB, acc[0][nt]);
      acc[1][nt] = wmma_bf16(ha1, pB, acc[1][nt]);
    }
    __syncthreads();
  }

  float gm = gamma[0];
  for (int mt = 0; mt < 2; ++mt)
    for (int nt = 0; nt < 4; ++nt)
      for (int r = 0; r < 8; ++r) {
        int c = 32 * w + 16 * mt + r + 8 * hi;
        int j = j0 + 16 * nt + l16;
        size_t o = ((size_t)b * CC + c) * NTOK + j;
        out[o] = gm * acc[mt][nt][r] + x[o];
      }
}

extern "C" void kernel_launch(void* const* d_in, const int* in_sizes, int n_in,
                              void* d_out, int out_size, void* d_ws, size_t ws_size,
                              hipStream_t stream) {
  const float* x     = (const float*)d_in[0];
  const float* Wf    = (const float*)d_in[1];
  const float* bf    = (const float*)d_in[2];
  const float* Wg    = (const float*)d_in[3];
  const float* bg    = (const float*)d_in[4];
  const float* Wh    = (const float*)d_in[5];
  const float* bh    = (const float*)d_in[6];
  const float* gamma = (const float*)d_in[7];
  float* out = (float*)d_out;

  char* p = (char*)d_ws;
  unsigned short* Wf_bf = (unsigned short*)p; p += (size_t)C8V * CC * 2;
  unsigned short* Wg_bf = (unsigned short*)p; p += (size_t)C8V * CC * 2;
  unsigned short* Wh_bf = (unsigned short*)p; p += (size_t)CC * CC * 2;
  unsigned short* f_ws  = (unsigned short*)p; p += (size_t)BB * NTOK * C8V * 2;
  unsigned short* gt_ws = (unsigned short*)p; p += (size_t)BB * NTOK * C8V * 2;
  unsigned short* hv_ws = (unsigned short*)p; p += (size_t)BB * CC * NTOK * 2;
  float* m_ws  = (float*)p; p += (size_t)BB * NTOK * 4;
  float* rl_ws = (float*)p; p += (size_t)BB * NTOK * 4;

  cvt_weights_kernel<<<dim3((CC * CC + 255) / 256), dim3(256), 0, stream>>>(
      Wf, Wg, Wh, Wf_bf, Wg_bf, Wh_bf);
  proj_kernel<<<dim3(NTOK / 64, BB), dim3(512), 0, stream>>>(
      x, bf, bg, bh, Wf_bf, Wg_bf, Wh_bf, f_ws, gt_ws, hv_ws);
  stats_kernel<<<dim3(NTOK / 256, BB), dim3(512), 0, stream>>>(
      f_ws, gt_ws, m_ws, rl_ws);
  attn_out_kernel<<<dim3(NTOK / 64, BB), dim3(512), 0, stream>>>(
      x, gamma, f_ws, gt_ws, hv_ws, m_ws, rl_ws, out);
}
